// PAM_15204184228135
// MI455X (gfx1250) — matrix operand order repaired and checked by reading, not compiled, not run on hardware
//
#include <hip/hip_runtime.h>

// ---------------------------------------------------------------------------
// PAM (parallax attention) for MI455X / gfx1250, wave32 + bf16 WMMA + TDM.
//
// - All GEMM activations NHWC bf16; every WMMA fragment is a contiguous
//   32B/lane load; C/D packs to contiguous b128 stores.
// - Conv GEMMs: 16(oc) x 32(pix) per wave = two independent accumulator
//   chains sharing one A-fragment (better WMMA overlap, half weight loads).
// - 1x1 convs: block stages its 64px x 64ch activation tile (contiguous
//   8 KB) into LDS via TENSOR_LOAD_TO_LDS (1-D descriptor), waits on
//   TENSORcnt, then all 8 waves consume B-fragments from LDS.
// ---------------------------------------------------------------------------

typedef __attribute__((ext_vector_type(16))) __bf16 v16bf;
typedef __attribute__((ext_vector_type(8)))  __bf16 v8bf;
typedef __attribute__((ext_vector_type(2)))  __bf16 v2bf;
typedef __attribute__((ext_vector_type(8)))  float  v8f;
typedef __attribute__((ext_vector_type(4)))  unsigned v4u;
typedef __attribute__((ext_vector_type(8)))  int    v8i;
typedef __attribute__((ext_vector_type(4)))  int    v4i;

#define HW   32768      // H*W
#define Wd   256
#define Hh   128
#define Cc   64
#define Bb   2

#if __has_builtin(__builtin_amdgcn_tensor_load_to_lds)
#define HAS_TDM 1
#else
#define HAS_TDM 0
#endif

static __device__ __forceinline__ v16bf vzero16() {
    union { v16bf v; unsigned u[8]; } z;
#pragma unroll
    for (int i = 0; i < 8; ++i) z.u[i] = 0u;
    return z.v;
}

#if HAS_TDM
// 1-D contiguous TDM copy: n8 units of 8 bytes, global -> LDS.
static __device__ __forceinline__ void tdm_load_1d(unsigned ldsoff,
                                                   const void* gptr,
                                                   unsigned n8) {
    unsigned long long ga = (unsigned long long)gptr;
    v4u g0 = { 1u,                                   // count=1, user desc
               ldsoff,                               // lds_addr (bytes)
               (unsigned)ga,                         // global_addr[31:0]
               (unsigned)((ga >> 32) & 0x1FFFFFFu) | (2u << 30) }; // +type=2
    v8i g1;
    g1[0] = 0x00030000;                              // data_size = 8B
    g1[1] = (int)((n8 & 0xFFFFu) << 16);             // tensor_dim0 lo16
    g1[2] = (int)(((n8 >> 16) & 0xFFFFu) | (1u << 16)); // dim0 hi | dim1=1
    g1[3] = (int)((n8 & 0xFFFFu) << 16);             // tile_dim0 (n8 < 64K)
    g1[4] = 0;                                       // tile_dim1/2 unused
    g1[5] = (int)n8;                                 // tensor_dim0_stride lo
    g1[6] = 0;
    g1[7] = 0;
    v4i z4 = {0, 0, 0, 0};
#if defined(__clang_major__) && __clang_major__ >= 23
    v8i z8 = {0, 0, 0, 0, 0, 0, 0, 0};
    __builtin_amdgcn_tensor_load_to_lds(g0, g1, z4, z4, z8, 0);
#else
    __builtin_amdgcn_tensor_load_to_lds(g0, g1, z4, z4, 0);
#endif
}
#endif

// ---------------------------------------------------------------------------
// NCHW fp32 -> NHWC bf16
// ---------------------------------------------------------------------------
__global__ __launch_bounds__(256) void k_nchw2nhwc_bf16(
    const float* __restrict__ src, __bf16* __restrict__ dst)
{
    int o   = blockIdx.x * 256 + threadIdx.x;   // B*HW*C = 4194304
    int c   = o & 63;
    int pix = o >> 6;
    int b   = pix >> 15;
    int p   = pix & (HW - 1);
    dst[o] = (__bf16)src[((size_t)(b * Cc + c) << 15) + p];
}

// ---------------------------------------------------------------------------
// Weight swizzle: A fragment (16x32 bf16): lane l -> m=l&15, sub=l>>4,
//   half i -> K = sub*16 + i
// ---------------------------------------------------------------------------
__global__ __launch_bounds__(256) void k_swz3x3(
    const float* __restrict__ w, __bf16* __restrict__ dst)
{
    int t    = blockIdx.x * 256 + threadIdx.x;  // 9*4*2*512 = 36864
    int i    = t & 15;
    int lane = (t >> 4) & 31;
    int f    = t >> 9;                          // ((tap*4 + octile)*2 + kc)
    int kc   = f & 1;
    int oct  = (f >> 1) & 3;
    int tap  = f >> 3;
    int ky   = tap / 3, kx = tap % 3;
    int KA   = (lane >> 4) * 16 + i;
    int oc   = oct * 16 + (lane & 15);
    int ic   = kc * 32 + KA;
    dst[t] = (__bf16)w[((oc * 64 + ic) * 3 + ky) * 3 + kx];
}

__global__ __launch_bounds__(256) void k_swz1x1(
    const float* __restrict__ w, __bf16* __restrict__ dst, int KC)
{
    int t    = blockIdx.x * 256 + threadIdx.x;  // 4*KC*512
    int i    = t & 15;
    int lane = (t >> 4) & 31;
    int f    = t >> 9;
    int kc   = f % KC;
    int oct  = f / KC;
    int KA   = (lane >> 4) * 16 + i;
    int oc   = oct * 16 + (lane & 15);
    int ic   = kc * 32 + KA;
    dst[t] = (__bf16)w[oc * (KC * 32) + ic];
}

// ---------------------------------------------------------------------------
// 3x3 conv: one wave = 16oc x 32pix (two sub-tiles, two accumulator chains,
// shared A fragment). 9 shifted GEMM accumulations, K = 64 ic in 2 chunks.
// mode 0: leaky_relu(0.1) -> NHWC bf16 ; mode 1: +residual(NCHW f32) -> bf16
// ---------------------------------------------------------------------------
__global__ __launch_bounds__(256) void k_conv3x3_wmma(
    const __bf16* __restrict__ in, const __bf16* __restrict__ wf,
    const float* __restrict__ resid, __bf16* __restrict__ out, int mode)
{
    int lane = threadIdx.x & 31;
    int gid  = blockIdx.x * 8 + (threadIdx.x >> 5);
    int oct  = gid & 3;
    int t    = gid >> 2;                // 0..2047
    int b    = t >> 10;                 // HW/32 = 1024 tiles per batch
    int pbase = (t & 1023) << 5;
    int nloc = lane & 15, sub = lane >> 4;
    int p0 = pbase + nloc;              // sub-tile 0 pixel
    int y  = p0 >> 8;                   // both sub-tiles share the row
    int x0 = p0 & 255;

    v8f acc0 = {}, acc1 = {};
#pragma unroll
    for (int ky = 0; ky < 3; ++ky) {
        int yy = y + ky - 1;
        bool yok = (yy >= 0) && (yy < Hh);
#pragma unroll
        for (int kx = 0; kx < 3; ++kx) {
            int xx0 = x0 + kx - 1;
            int xx1 = xx0 + 16;
            bool ok0 = yok && (xx0 >= 0) && (xx0 < Wd);
            bool ok1 = yok && (xx1 < Wd);          // xx1 >= 15 always
            const __bf16* bp0 =
                in + ((size_t)(b << 15) + (yy << 8) + xx0) * Cc + sub * 16;
#pragma unroll
            for (int kc = 0; kc < 2; ++kc) {
                v16bf bf0 = ok0 ? *(const v16bf*)(bp0 + kc * 32) : vzero16();
                v16bf bf1 = ok1 ? *(const v16bf*)(bp0 + 16 * Cc + kc * 32)
                                : vzero16();
                int f = ((ky * 3 + kx) * 4 + oct) * 2 + kc;
                v16bf af = *(const v16bf*)(wf + f * 512 + lane * 16);
                acc0 = __builtin_amdgcn_wmma_f32_16x16x32_bf16(
                    false, af, false, bf0, (short)0, acc0, false, false);
                acc1 = __builtin_amdgcn_wmma_f32_16x16x32_bf16(
                    false, af, false, bf1, (short)0, acc1, false, false);
            }
        }
    }

    int oc0 = oct * 16 + sub * 8;
#pragma unroll
    for (int half = 0; half < 2; ++half) {
        v8f acc = half ? acc1 : acc0;
        int p = p0 + half * 16;
        if (mode == 0) {
#pragma unroll
            for (int r = 0; r < 8; ++r) {
                float v = acc[r];
                acc[r] = v > 0.0f ? v : 0.1f * v;
            }
        } else {
            const float* rp = resid + ((size_t)(b * Cc + oc0) << 15) + p;
#pragma unroll
            for (int r = 0; r < 8; ++r) acc[r] += rp[(size_t)r << 15];
        }
        v8bf ob;
#pragma unroll
        for (int r = 0; r < 8; ++r) ob[r] = (__bf16)acc[r];
        *(v8bf*)(out + ((size_t)(b << 15) + p) * Cc + oc0) = ob;
    }
}

// ---------------------------------------------------------------------------
// 1x1 conv GEMM. Block = 64 consecutive pixels; activation tile(s) staged
// into LDS once (TDM if available), shared by 4 octile x 2 pixel-group waves.
// mode 0: store fp32 NHWC; mode 1: store fp32 NCHW (final output).
// ---------------------------------------------------------------------------
__global__ __launch_bounds__(256) void k_conv1x1_wmma(
    const __bf16* __restrict__ src0, const __bf16* __restrict__ src1, int KC,
    const __bf16* __restrict__ wf, const float* __restrict__ bias,
    float* __restrict__ outNHWC, float* __restrict__ outNCHW, int mode,
    int touch)
{
    __shared__ __bf16 tile[2][64 * 64];     // 2 x 8 KB NHWC tiles

    int lane = threadIdx.x & 31;
    int wid  = threadIdx.x >> 5;
    int blk  = blockIdx.x;                  // 1024 blocks
    int b    = blk >> 9;                    // 512 blocks per batch
    int pb   = (blk & 511) << 6;            // 64-pixel base
    size_t pixbase = (size_t)(b << 15) + pb;

    if (touch) {                            // never taken; keeps LDS "written"
        tile[0][threadIdx.x] = (__bf16)0.0f;
        tile[1][threadIdx.x] = (__bf16)0.0f;
    }

#if HAS_TDM
    if (wid == 0) {
        unsigned lds0 = (unsigned)(unsigned long long)&tile[0][0];
        tdm_load_1d(lds0, src0 + pixbase * Cc, 1024);       // 8 KB
        if (KC == 4) {
            unsigned lds1 = (unsigned)(unsigned long long)&tile[1][0];
            tdm_load_1d(lds1, src1 + pixbase * Cc, 1024);
        }
        __builtin_amdgcn_s_wait_tensorcnt(0);
    }
#else
    {   // cooperative fallback: 256 threads x 16 bytes x 2 iters = 8 KB
        const uint4* g0 = (const uint4*)(src0 + pixbase * Cc);
        uint4* s0 = (uint4*)&tile[0][0];
#pragma unroll
        for (int it = 0; it < 2; ++it)
            s0[it * 256 + threadIdx.x] = g0[it * 256 + threadIdx.x];
        if (KC == 4) {
            const uint4* g1 = (const uint4*)(src1 + pixbase * Cc);
            uint4* s1 = (uint4*)&tile[1][0];
#pragma unroll
            for (int it = 0; it < 2; ++it)
                s1[it * 256 + threadIdx.x] = g1[it * 256 + threadIdx.x];
        }
    }
#endif
    __syncthreads();

    int oct = wid & 3;
    int grp = wid >> 2;                     // 0..1 -> 32-pixel group
    int nloc = lane & 15, sub = lane >> 4;
    int lp0 = grp * 32 + nloc;              // local pixels in [0,64)

    v8f acc0 = {}, acc1 = {};
    for (int kc = 0; kc < KC; ++kc) {
        int ch  = kc * 32 + sub * 16;       // 0..127, never straddles 64
        int sel = ch >> 6;
        int cl  = ch & 63;
        v16bf bf0 = *(const v16bf*)&tile[sel][lp0 * 64 + cl];
        v16bf bf1 = *(const v16bf*)&tile[sel][(lp0 + 16) * 64 + cl];
        v16bf af  = *(const v16bf*)(wf + (oct * KC + kc) * 512 + lane * 16);
        acc0 = __builtin_amdgcn_wmma_f32_16x16x32_bf16(
            false, af, false, bf0, (short)0, acc0, false, false);
        acc1 = __builtin_amdgcn_wmma_f32_16x16x32_bf16(
            false, af, false, bf1, (short)0, acc1, false, false);
    }

    int oc0 = oct * 16 + sub * 8;
#pragma unroll
    for (int half = 0; half < 2; ++half) {
        v8f acc = half ? acc1 : acc0;
#pragma unroll
        for (int r = 0; r < 8; ++r) acc[r] += bias[oc0 + r];
        size_t pix = pixbase + lp0 + half * 16;
        if (mode == 0) {
            *(v8f*)(outNHWC + pix * Cc + oc0) = acc;
        } else {
            int p = (int)(pix & (HW - 1));
#pragma unroll
            for (int r = 0; r < 8; ++r)
                outNCHW[((size_t)(b * Cc + oc0 + r) << 15) + p] = acc[r];
        }
    }
}

// ---------------------------------------------------------------------------
// Attention: one wave per (b, p). lane=candidate for score/softmax (shfl
// reductions), lane=channel-pair for value accumulation (coalesced gathers).
// ---------------------------------------------------------------------------
__global__ __launch_bounds__(256) void k_attention(
    const float* __restrict__ Q, const float* __restrict__ S,
    const float* __restrict__ R, const int* __restrict__ Px,
    const int* __restrict__ Py, __bf16* __restrict__ buf)
{
    int lane = threadIdx.x & 31;
    int gid  = blockIdx.x * 8 + (threadIdx.x >> 5);
    int b    = gid >> 15;
    int p    = gid & (HW - 1);

    int px  = Px[p * 32 + lane];
    int py  = Py[p * 32 + lane];
    int pos = px * Wd + py;

    const float4* q4 = (const float4*)(Q + ((size_t)(b << 15) + p) * Cc);
    const float4* s4 = (const float4*)(S + ((size_t)(b << 15) + pos) * Cc);
    float sc = 0.0f;
#pragma unroll
    for (int i = 0; i < 16; ++i) {
        float4 a = q4[i], s = s4[i];
        sc += a.x * s.x + a.y * s.y + a.z * s.z + a.w * s.w;
    }

    float m = sc;
#pragma unroll
    for (int off = 16; off > 0; off >>= 1) m = fmaxf(m, __shfl_xor(m, off));
    float e = __expf(sc - m);
    float sum = e;
#pragma unroll
    for (int off = 16; off > 0; off >>= 1) sum += __shfl_xor(sum, off);
    float attn = e / sum;

    float acc0 = 0.0f, acc1 = 0.0f;
#pragma unroll 4
    for (int kk = 0; kk < 32; ++kk) {
        float a = __shfl(attn, kk);
        int  pp = __shfl(pos, kk);
        const float2* rv =
            (const float2*)(R + ((size_t)(b << 15) + pp) * Cc) + lane;
        float2 v = *rv;
        acc0 += a * v.x;
        acc1 += a * v.y;
    }
    v2bf ov;
    ov[0] = (__bf16)acc0;
    ov[1] = (__bf16)acc1;
    *(v2bf*)(buf + ((size_t)(b << 15) + p) * Cc + lane * 2) = ov;
}

// ---------------------------------------------------------------------------
// Workspace layout (bytes). One activation tensor (NHWC bf16) = 8 MB.
// ---------------------------------------------------------------------------
#define A_BYTES  ((size_t)8388608)
#define F_BYTES  ((size_t)16777216)
#define O_XL     ((size_t)0)
#define O_XR     (A_BYTES)
#define O_HDN    (2 * A_BYTES)
#define O_BUFL   (3 * A_BYTES)
#define O_BUFR   (4 * A_BYTES)
#define O_BUFATT (5 * A_BYTES)
#define O_Q      (6 * A_BYTES)
#define O_S      (O_Q + F_BYTES)
#define O_R      (O_S + F_BYTES)
#define O_WR1    (O_R + F_BYTES)
#define O_WR2    (O_WR1 + (size_t)73728)
#define O_WB1    (O_WR1 + (size_t)147456)
#define O_WB2    (O_WB1 + (size_t)8192)
#define O_WB3    (O_WB2 + (size_t)8192)
#define O_WFUS   (O_WB3 + (size_t)8192)

extern "C" void kernel_launch(void* const* d_in, const int* in_sizes, int n_in,
                              void* d_out, int out_size, void* d_ws, size_t ws_size,
                              hipStream_t stream) {
    const float* x_left  = (const float*)d_in[0];
    const float* x_right = (const float*)d_in[1];
    const int*   Pos_x   = (const int*)d_in[2];
    const int*   Pos_y   = (const int*)d_in[3];
    // d_in[4] = is_training (inference path only)
    const float* rb_w1 = (const float*)d_in[5];
    const float* rb_w2 = (const float*)d_in[6];
    const float* b1_w  = (const float*)d_in[7];
    const float* b1_b  = (const float*)d_in[8];
    const float* b2_w  = (const float*)d_in[9];
    const float* b2_b  = (const float*)d_in[10];
    const float* b3_w  = (const float*)d_in[11];
    const float* b3_b  = (const float*)d_in[12];
    const float* fus_w = (const float*)d_in[13];
    const float* fus_b = (const float*)d_in[14];
    float* out = (float*)d_out;

    char* ws = (char*)d_ws;
    __bf16* xl_bf  = (__bf16*)(ws + O_XL);
    __bf16* xr_bf  = (__bf16*)(ws + O_XR);
    __bf16* hdn    = (__bf16*)(ws + O_HDN);
    __bf16* bufl   = (__bf16*)(ws + O_BUFL);
    __bf16* bufr   = (__bf16*)(ws + O_BUFR);
    __bf16* bufatt = (__bf16*)(ws + O_BUFATT);
    float*  Q      = (float*)(ws + O_Q);
    float*  S      = (float*)(ws + O_S);
    float*  R      = (float*)(ws + O_R);
    __bf16* wr1  = (__bf16*)(ws + O_WR1);
    __bf16* wr2  = (__bf16*)(ws + O_WR2);
    __bf16* wb1  = (__bf16*)(ws + O_WB1);
    __bf16* wb2  = (__bf16*)(ws + O_WB2);
    __bf16* wb3  = (__bf16*)(ws + O_WB3);
    __bf16* wfus = (__bf16*)(ws + O_WFUS);

    // --- stage 0: layout conversion + weight swizzle ---
    k_nchw2nhwc_bf16<<<16384, 256, 0, stream>>>(x_left,  xl_bf);
    k_nchw2nhwc_bf16<<<16384, 256, 0, stream>>>(x_right, xr_bf);
    k_swz3x3<<<144, 256, 0, stream>>>(rb_w1, wr1);
    k_swz3x3<<<144, 256, 0, stream>>>(rb_w2, wr2);
    k_swz1x1<<<16, 256, 0, stream>>>(b1_w, wb1, 2);
    k_swz1x1<<<16, 256, 0, stream>>>(b2_w, wb2, 2);
    k_swz1x1<<<16, 256, 0, stream>>>(b3_w, wb3, 2);
    k_swz1x1<<<32, 256, 0, stream>>>(fus_w, wfus, 4);

    // --- stage 1: residual blocks (left, then right, reusing hdn) ---
    k_conv3x3_wmma<<<1024, 256, 0, stream>>>(xl_bf, wr1, nullptr, hdn, 0);
    k_conv3x3_wmma<<<1024, 256, 0, stream>>>(hdn, wr2, x_left, bufl, 1);
    k_conv3x3_wmma<<<1024, 256, 0, stream>>>(xr_bf, wr1, nullptr, hdn, 0);
    k_conv3x3_wmma<<<1024, 256, 0, stream>>>(hdn, wr2, x_right, bufr, 1);

    // --- stage 2: Q / S / R projections (fp32 NHWC for attention) ---
    k_conv1x1_wmma<<<1024, 256, 0, stream>>>(bufl, nullptr, 2, wb1, b1_b, Q, nullptr, 0, 0);
    k_conv1x1_wmma<<<1024, 256, 0, stream>>>(bufr, nullptr, 2, wb2, b2_b, S, nullptr, 0, 0);
    k_conv1x1_wmma<<<1024, 256, 0, stream>>>(xr_bf, nullptr, 2, wb3, b3_b, R, nullptr, 0, 0);

    // --- stage 3: gather + softmax attention -> buf (NHWC bf16) ---
    k_attention<<<8192, 256, 0, stream>>>(Q, S, R, Pos_x, Pos_y, bufatt);

    // --- stage 4: fusion conv over concat(buf, x_left) -> NCHW fp32 out ---
    k_conv1x1_wmma<<<1024, 256, 0, stream>>>(bufatt, xl_bf, 4, wfus, fus_b,
                                             nullptr, out, 1, 0);
}
